// MoETransformerBlock_85143431675951
// MI455X (gfx1250) — matrix operand order repaired and checked by reading, not compiled, not run on hardware
//
#include <hip/hip_runtime.h>

// ---------------- problem constants (from reference) ----------------
constexpr int Bc = 2, Sc = 2048, Dc = 1024;
constexpr int Hc = 16, KVHc = 4, HDc = 64;
constexpr int Ec = 8, TOPKc = 2, Fc = 4096;
constexpr int Tc = Bc * Sc;                 // 4096 tokens

typedef __bf16 bf16;
typedef __attribute__((ext_vector_type(16))) __bf16 v16bf;
typedef __attribute__((ext_vector_type(8)))  __bf16 v8bf;
typedef __attribute__((ext_vector_type(2)))  __bf16 v2bf;
typedef __attribute__((ext_vector_type(8)))  float  v8f;
typedef __attribute__((ext_vector_type(4)))  unsigned int u32x4;
typedef __attribute__((ext_vector_type(8)))  int i32x8;
typedef __attribute__((ext_vector_type(4)))  int i32x4;

// ---------------- GEMM tiling ----------------
#define TILE_M 128
#define TILE_N 64
#define TILE_K 32

// ======================================================================
// Tensor Data Mover: DMA a [TILE_M x TILE_K] bf16 tile (row stride K
// elements) from global into LDS, packed contiguously (64B rows) —
// exactly the sA[TILE_M][TILE_K] layout. D# per CDNA5 ISA §8.
// Issued once per wave (EXEC-independent); tracked by TENSORcnt.
// This toolchain uses the 6-arg builtin:
//   (u32x4 g0, i32x8 g1, i32x4 g2, i32x4 g3, i32x8 extra, i32 cpol)
// ======================================================================
__device__ __forceinline__ void tdm_load_a_tile(const bf16* gsrc,
                                                unsigned lds_addr,
                                                int K, int M)
{
    const unsigned long long ga = (unsigned long long)(uintptr_t)gsrc;
    u32x4 g0;
    g0[0] = 1u;                                            // count=1, user D#, no gather
    g0[1] = lds_addr;                                      // LDS byte address
    g0[2] = (unsigned)ga;                                  // global_addr[31:0]
    g0[3] = (unsigned)((ga >> 32) & 0x01FFFFFFu)           // global_addr[56:32]
          | (2u << 30);                                    // type = 2 ("image")
    i32x8 g1;
    g1[0] = (int)(1u << 16);                               // data_size=1 (2 bytes)
    g1[1] = (int)(((unsigned)K & 0xFFFFu) << 16);          // tensor_dim0[15:0]
    g1[2] = (int)(((unsigned)K >> 16)                      // tensor_dim0[31:16]
          | (((unsigned)M & 0xFFFFu) << 16));              // tensor_dim1[15:0]
    g1[3] = (int)(((unsigned)M >> 16)                      // tensor_dim1[31:16]
          | ((unsigned)TILE_K << 16));                     // tile_dim0 = 32
    g1[4] = TILE_M;                                        // tile_dim1 = 128, tile_dim2 = 0
    g1[5] = K;                                             // tensor_dim0_stride[31:0]
    g1[6] = 0;                                             // stride hi / dim1_stride lo
    g1[7] = 0;
    const i32x4 gz4 = {0, 0, 0, 0};                        // 2D: groups 2/3 unused
    const i32x8 gz8 = {0, 0, 0, 0, 0, 0, 0, 0};
    __builtin_amdgcn_tensor_load_to_lds(g0, g1, gz4, gz4, gz8, 0);
}

// ======================================================================
// RMSNorm: f32 [rows, Dc] -> bf16 [rows, Dc]
// ======================================================================
__global__ __launch_bounds__(256) void rmsnorm_kernel(
    const float* __restrict__ x, const float* __restrict__ w,
    bf16* __restrict__ out)
{
    __shared__ float red[256];
    const size_t row = blockIdx.x;
    const float* xr = x + row * Dc;
    float s = 0.f;
    for (int d = threadIdx.x; d < Dc; d += 256) { float v = xr[d]; s += v * v; }
    red[threadIdx.x] = s; __syncthreads();
    for (int st = 128; st > 0; st >>= 1) {
        if (threadIdx.x < st) red[threadIdx.x] += red[threadIdx.x + st];
        __syncthreads();
    }
    const float rs = rsqrtf(red[0] / (float)Dc + 1e-6f);
    for (int d = threadIdx.x; d < Dc; d += 256)
        out[row * Dc + d] = (bf16)(xr[d] * rs * w[d]);
}

// ======================================================================
// bf16 WMMA GEMM: C[M,N] (f32) = A[M,K](bf16) * W[K,N](f32 cast->bf16)
//   + optional residual[M,N], optional per-row scale, optional C-accum.
// A tile staged by the TDM (async DMA to LDS, TENSORcnt); W tile staged
// by VALU with packed bf16 dword stores. 8 waves; wave computes a 32x32
// patch as 2x2 v_wmma_f32_16x16x32_bf16 tiles.
// ======================================================================
__global__ __launch_bounds__(256) void gemm_bf16_kernel(
    const bf16*  __restrict__ A, const float* __restrict__ W,
    float* __restrict__ C,
    const float* __restrict__ residual,
    const float* __restrict__ rowscale, int rs_stride,
    int M, int N, int K, int accumulate)
{
    __shared__ bf16 sA [TILE_M][TILE_K];   // 8 KB, K-contiguous rows (TDM dest)
    __shared__ bf16 sBT[TILE_N][TILE_K];   // 4 KB, B transposed (K-contiguous)

    const int tid  = threadIdx.x;
    const int lane = tid & 31;
    const int wave = tid >> 5;
    const int wm   = wave >> 1;            // 0..3
    const int wn   = wave & 1;             // 0..1
    const int m0   = blockIdx.y * TILE_M;
    const int n0   = blockIdx.x * TILE_N;
    const int lhalf = lane >> 4;           // 0/1
    const int lmod  = lane & 15;
    const unsigned sA_lds = (unsigned)(uintptr_t)&sA[0][0];

    v8f acc[2][2];
    #pragma unroll
    for (int i = 0; i < 2; ++i)
        #pragma unroll
        for (int j = 0; j < 2; ++j)
            #pragma unroll
            for (int r = 0; r < 8; ++r) acc[i][j][r] = 0.f;

    for (int k0 = 0; k0 < K; k0 += TILE_K) {
        // ---- A tile via Tensor Data Mover (one wave issues the DMA) ----
        if (wave == 0)
            tdm_load_a_tile(A + (size_t)m0 * K + k0, sA_lds, K, M);

        // ---- W tile: f32 -> packed bf16 dwords, K-transposed ----
        {
            const int kp = tid >> 4;          // 0..15 (pairs of k)
            const int nc = (tid & 15) * 4;    // 0,4,...,60
            const float* s0 = W + (size_t)(k0 + 2 * kp) * N + n0 + nc;
            const float* s1 = s0 + N;
            #pragma unroll
            for (int j = 0; j < 4; ++j) {
                v2bf p; p[0] = (bf16)s0[j]; p[1] = (bf16)s1[j];
                *(v2bf*)&sBT[nc + j][2 * kp] = p;
            }
            if (k0 + TILE_K < K)  // gfx1250 global_prefetch of next W slab
                __builtin_prefetch(s0 + (size_t)TILE_K * N, 0, 1);
        }
        if (wave == 0) __builtin_amdgcn_s_wait_tensorcnt(0);
        __syncthreads();

        // ---- build fragments per ISA 7.12.2 layouts ----
        v16bf afrag[2], bfrag[2];
        #pragma unroll
        for (int i = 0; i < 2; ++i) {
            const int m = wm * 32 + i * 16 + lmod;
            v8bf lo = *(const v8bf*)&sA[m][lhalf * 8];
            v8bf hi = *(const v8bf*)&sA[m][16 + lhalf * 8];
            v16bf f;
            #pragma unroll
            for (int e = 0; e < 8; ++e) { f[e] = lo[e]; f[e + 8] = hi[e]; }
            afrag[i] = f;
        }
        #pragma unroll
        for (int j = 0; j < 2; ++j) {
            const int n = wn * 32 + j * 16 + lmod;
            v8bf lo = *(const v8bf*)&sBT[n][lhalf * 8];
            v8bf hi = *(const v8bf*)&sBT[n][16 + lhalf * 8];
            v16bf f;
            #pragma unroll
            for (int e = 0; e < 8; ++e) { f[e] = lo[e]; f[e + 8] = hi[e]; }
            bfrag[j] = f;
        }
        #pragma unroll
        for (int i = 0; i < 2; ++i)
            #pragma unroll
            for (int j = 0; j < 2; ++j)
                acc[i][j] = __builtin_amdgcn_wmma_f32_16x16x32_bf16(
                    false, afrag[i], false, bfrag[j],
                    (short)0, acc[i][j], false, false);
        __syncthreads();
    }

    // ---- epilogue: C/D layout — VGPR r: lanes0-15 M=r, lanes16-31 M=8+r ----
    #pragma unroll
    for (int i = 0; i < 2; ++i)
        #pragma unroll
        for (int j = 0; j < 2; ++j)
            #pragma unroll
            for (int r = 0; r < 8; ++r) {
                const int m = m0 + wm * 32 + i * 16 + lhalf * 8 + r;
                const int n = n0 + wn * 32 + j * 16 + lmod;
                float v = acc[i][j][r];
                if (rowscale)  v *= rowscale[(size_t)m * rs_stride];
                const size_t off = (size_t)m * N + n;
                if (residual)  v += residual[off];
                if (accumulate) v += C[off];
                C[off] = v;
            }
}

// ======================================================================
// Fused MoE gate+up GEMM with SwiGLU epilogue -> bf16 activations
// Act[M,N] = bf16( silu(A*Wg) * (A*Wu) ).  A tile via TDM.
// ======================================================================
__global__ __launch_bounds__(256) void moe_gateup_kernel(
    const bf16* __restrict__ A,
    const float* __restrict__ Wg, const float* __restrict__ Wu,
    bf16* __restrict__ Act, int M, int N, int K)
{
    __shared__ bf16 sA  [TILE_M][TILE_K];
    __shared__ bf16 sBTg[TILE_N][TILE_K];
    __shared__ bf16 sBTu[TILE_N][TILE_K];

    const int tid  = threadIdx.x;
    const int lane = tid & 31;
    const int wave = tid >> 5;
    const int wm   = wave >> 1;
    const int wn   = wave & 1;
    const int m0   = blockIdx.y * TILE_M;
    const int n0   = blockIdx.x * TILE_N;
    const int lhalf = lane >> 4;
    const int lmod  = lane & 15;
    const unsigned sA_lds = (unsigned)(uintptr_t)&sA[0][0];

    v8f accg[2][2], accu[2][2];
    #pragma unroll
    for (int i = 0; i < 2; ++i)
        #pragma unroll
        for (int j = 0; j < 2; ++j)
            #pragma unroll
            for (int r = 0; r < 8; ++r) { accg[i][j][r] = 0.f; accu[i][j][r] = 0.f; }

    for (int k0 = 0; k0 < K; k0 += TILE_K) {
        if (wave == 0)
            tdm_load_a_tile(A + (size_t)m0 * K + k0, sA_lds, K, M);
        {
            const int kp = tid >> 4;
            const int nc = (tid & 15) * 4;
            const float* g0p = Wg + (size_t)(k0 + 2 * kp) * N + n0 + nc;
            const float* g1p = g0p + N;
            const float* u0p = Wu + (size_t)(k0 + 2 * kp) * N + n0 + nc;
            const float* u1p = u0p + N;
            #pragma unroll
            for (int j = 0; j < 4; ++j) {
                v2bf pg; pg[0] = (bf16)g0p[j]; pg[1] = (bf16)g1p[j];
                v2bf pu; pu[0] = (bf16)u0p[j]; pu[1] = (bf16)u1p[j];
                *(v2bf*)&sBTg[nc + j][2 * kp] = pg;
                *(v2bf*)&sBTu[nc + j][2 * kp] = pu;
            }
            if (k0 + TILE_K < K) {
                __builtin_prefetch(g0p + (size_t)TILE_K * N, 0, 1);
                __builtin_prefetch(u0p + (size_t)TILE_K * N, 0, 1);
            }
        }
        if (wave == 0) __builtin_amdgcn_s_wait_tensorcnt(0);
        __syncthreads();

        v16bf afrag[2], bgfrag[2], bufrag[2];
        #pragma unroll
        for (int i = 0; i < 2; ++i) {
            const int m = wm * 32 + i * 16 + lmod;
            v8bf lo = *(const v8bf*)&sA[m][lhalf * 8];
            v8bf hi = *(const v8bf*)&sA[m][16 + lhalf * 8];
            v16bf f;
            #pragma unroll
            for (int e = 0; e < 8; ++e) { f[e] = lo[e]; f[e + 8] = hi[e]; }
            afrag[i] = f;
        }
        #pragma unroll
        for (int j = 0; j < 2; ++j) {
            const int n = wn * 32 + j * 16 + lmod;
            v8bf glo = *(const v8bf*)&sBTg[n][lhalf * 8];
            v8bf ghi = *(const v8bf*)&sBTg[n][16 + lhalf * 8];
            v8bf ulo = *(const v8bf*)&sBTu[n][lhalf * 8];
            v8bf uhi = *(const v8bf*)&sBTu[n][16 + lhalf * 8];
            v16bf fg, fu;
            #pragma unroll
            for (int e = 0; e < 8; ++e) {
                fg[e] = glo[e]; fg[e + 8] = ghi[e];
                fu[e] = ulo[e]; fu[e + 8] = uhi[e];
            }
            bgfrag[j] = fg; bufrag[j] = fu;
        }
        #pragma unroll
        for (int i = 0; i < 2; ++i)
            #pragma unroll
            for (int j = 0; j < 2; ++j) {
                accg[i][j] = __builtin_amdgcn_wmma_f32_16x16x32_bf16(
                    false, afrag[i], false, bgfrag[j], (short)0, accg[i][j], false, false);
                accu[i][j] = __builtin_amdgcn_wmma_f32_16x16x32_bf16(
                    false, afrag[i], false, bufrag[j], (short)0, accu[i][j], false, false);
            }
        __syncthreads();
    }

    #pragma unroll
    for (int i = 0; i < 2; ++i)
        #pragma unroll
        for (int j = 0; j < 2; ++j)
            #pragma unroll
            for (int r = 0; r < 8; ++r) {
                const int m = m0 + wm * 32 + i * 16 + lhalf * 8 + r;
                const int n = n0 + wn * 32 + j * 16 + lmod;
                const float g = accg[i][j][r];
                const float u = accu[i][j][r];
                const float silu = g / (1.f + __expf(-g));
                Act[(size_t)m * N + n] = (bf16)(silu * u);
            }
}

// ======================================================================
// RoPE (HF rotate-half), in place on f32 [Tc, heads, 64]
// ======================================================================
__global__ __launch_bounds__(256) void rope_kernel(float* __restrict__ x,
                                                   int heads, int total)
{
    const int idx = blockIdx.x * 256 + threadIdx.x;
    if (idx >= total) return;
    const int i    = idx & 31;
    const int head = (idx >> 5) % heads;
    const int t    = idx / (32 * heads);
    const int s    = t % Sc;
    const float inv_freq = powf(10000.0f, -((float)(2 * i)) / 64.0f);
    const float ang = (float)s * inv_freq;
    const float c = cosf(ang), sn = sinf(ang);
    float* base = x + ((size_t)t * heads + head) * 64;
    const float x1 = base[i], x2 = base[i + 32];
    base[i]      = x1 * c - x2 * sn;
    base[i + 32] = x2 * c + x1 * sn;
}

// ======================================================================
// Causal attention, online softmax. One wave32 per query row (b,h,s);
// lane holds 2 of the 64 head-dim elements; K/V streamed (L2-resident).
// ======================================================================
__global__ __launch_bounds__(256) void attention_kernel(
    const float* __restrict__ q, const float* __restrict__ k,
    const float* __restrict__ v, bf16* __restrict__ ctx)
{
    const int lane = threadIdx.x & 31;
    const int row  = blockIdx.x * 8 + (threadIdx.x >> 5);   // (b*Hc + h)*Sc + s
    const int b   = row / (Hc * Sc);
    const int rem = row % (Hc * Sc);
    const int h   = rem / Sc;
    const int s   = rem % Sc;
    const int kvh = h / (Hc / KVHc);

    const float* qp = q + ((size_t)(b * Sc + s) * Hc + h) * HDc;
    const float q0 = qp[2 * lane], q1 = qp[2 * lane + 1];

    const float* kp = k + ((size_t)b * Sc * KVHc + kvh) * HDc;
    const float* vp = v + ((size_t)b * Sc * KVHc + kvh) * HDc;

    float m_ = -3.4e38f, l_ = 0.f, a0 = 0.f, a1 = 0.f;
    for (int t = 0; t <= s; ++t) {
        float p = q0 * kp[2 * lane] + q1 * kp[2 * lane + 1];
        #pragma unroll
        for (int off = 16; off > 0; off >>= 1) p += __shfl_xor(p, off, 32);
        p *= 0.125f;                               // 1/sqrt(64)
        const float nm = fmaxf(m_, p);
        const float sc = __expf(m_ - nm);
        const float w  = __expf(p - nm);
        l_ = l_ * sc + w;
        a0 = a0 * sc + w * vp[2 * lane];
        a1 = a1 * sc + w * vp[2 * lane + 1];
        m_ = nm;
        kp += KVHc * HDc; vp += KVHc * HDc;
    }
    const float inv = 1.f / l_;
    bf16* cp = ctx + ((size_t)(b * Sc + s) * Hc + h) * HDc;
    cp[2 * lane]     = (bf16)(a0 * inv);
    cp[2 * lane + 1] = (bf16)(a1 * inv);
}

// ======================================================================
// Router: one wave per token. logits -> softmax -> top-2 -> renorm gates.
// ======================================================================
__global__ __launch_bounds__(32) void router_kernel(
    const bf16* __restrict__ h2, const float* __restrict__ rw,
    float* __restrict__ gates, float* __restrict__ probs)
{
    const int t = blockIdx.x;
    const int lane = threadIdx.x;
    const bf16* xr = h2 + (size_t)t * Dc;

    float acc[Ec];
    #pragma unroll
    for (int e = 0; e < Ec; ++e) acc[e] = 0.f;
    for (int d = lane; d < Dc; d += 32) {
        const float xv = (float)xr[d];
        #pragma unroll
        for (int e = 0; e < Ec; ++e) acc[e] += xv * rw[(size_t)e * Dc + d];
    }
    #pragma unroll
    for (int e = 0; e < Ec; ++e)
        #pragma unroll
        for (int off = 16; off > 0; off >>= 1) acc[e] += __shfl_xor(acc[e], off, 32);

    if (lane == 0) {
        float mx = acc[0];
        #pragma unroll
        for (int e = 1; e < Ec; ++e) mx = fmaxf(mx, acc[e]);
        float p[Ec]; float sum = 0.f;
        #pragma unroll
        for (int e = 0; e < Ec; ++e) { p[e] = __expf(acc[e] - mx); sum += p[e]; }
        const float inv = 1.f / sum;
        #pragma unroll
        for (int e = 0; e < Ec; ++e) p[e] *= inv;
        int i1 = 0;
        #pragma unroll
        for (int e = 1; e < Ec; ++e) if (p[e] > p[i1]) i1 = e;
        int i2 = (i1 == 0) ? 1 : 0;
        #pragma unroll
        for (int e = 0; e < Ec; ++e) if (e != i1 && p[e] > p[i2]) i2 = e;
        const float renorm = 1.f / (p[i1] + p[i2]);
        #pragma unroll
        for (int e = 0; e < Ec; ++e) {
            probs[(size_t)t * Ec + e] = p[e];
            gates[(size_t)t * Ec + e] = 0.f;
        }
        gates[(size_t)t * Ec + i1] = p[i1] * renorm;
        gates[(size_t)t * Ec + i2] = p[i2] * renorm;
    }
}

// ======================================================================
// Balance loss: deterministic tree reduction (no atomics).
// ======================================================================
__global__ __launch_bounds__(256) void balance_kernel(
    const float* __restrict__ gates, const float* __restrict__ probs,
    float* __restrict__ loss_out)
{
    __shared__ float red[256];
    const int tid = threadIdx.x;
    float loss = 0.f;
    for (int e = 0; e < Ec; ++e) {
        float ps = 0.f, cs = 0.f;
        for (int t = tid; t < Tc; t += 256) {
            ps += probs[(size_t)t * Ec + e];
            cs += (gates[(size_t)t * Ec + e] > 0.f) ? 1.f : 0.f;
        }
        red[tid] = ps; __syncthreads();
        for (int st = 128; st > 0; st >>= 1) { if (tid < st) red[tid] += red[tid + st]; __syncthreads(); }
        const float psum = red[0]; __syncthreads();
        red[tid] = cs; __syncthreads();
        for (int st = 128; st > 0; st >>= 1) { if (tid < st) red[tid] += red[tid + st]; __syncthreads(); }
        const float csum = red[0]; __syncthreads();
        loss += (csum / (float)(Tc * TOPKc)) * (psum / (float)Tc);
    }
    if (tid == 0) *loss_out = 0.01f * (float)Ec * loss;
}

// ======================================================================
// Workspace layout (bytes) — total ~80.25 MiB, all re-written each call.
// ======================================================================
constexpr size_t OFF_H1    = 0;
constexpr size_t SZ_H1     = (size_t)Tc * Dc * 2;            //  8 MiB bf16
constexpr size_t OFF_Q     = OFF_H1 + SZ_H1;
constexpr size_t SZ_Q      = (size_t)Tc * Hc * HDc * 4;      // 16 MiB f32
constexpr size_t OFF_K     = OFF_Q + SZ_Q;
constexpr size_t SZ_KV     = (size_t)Tc * KVHc * HDc * 4;    //  4 MiB f32
constexpr size_t OFF_V     = OFF_K + SZ_KV;
constexpr size_t OFF_CTX   = OFF_V + SZ_KV;
constexpr size_t SZ_CTX    = (size_t)Tc * Hc * HDc * 2;      //  8 MiB bf16
constexpr size_t OFF_H2    = OFF_CTX + SZ_CTX;
constexpr size_t SZ_H2     = SZ_H1;                          //  8 MiB bf16
constexpr size_t OFF_GATES = OFF_H2 + SZ_H2;
constexpr size_t SZ_GE     = (size_t)Tc * Ec * 4;            // 128 KiB
constexpr size_t OFF_PROBS = OFF_GATES + SZ_GE;
constexpr size_t OFF_ACT   = OFF_PROBS + SZ_GE;
constexpr size_t SZ_ACT    = (size_t)Tc * Fc * 2;            // 32 MiB bf16

extern "C" void kernel_launch(void* const* d_in, const int* in_sizes, int n_in,
                              void* d_out, int out_size, void* d_ws, size_t ws_size,
                              hipStream_t stream)
{
    const float* x    = (const float*)d_in[0];   // hidden_states [B,S,D]
    const float* ln1w = (const float*)d_in[1];
    const float* wq   = (const float*)d_in[2];   // [D, H*HD]
    const float* wk   = (const float*)d_in[3];   // [D, KVH*HD]
    const float* wv   = (const float*)d_in[4];
    const float* wo   = (const float*)d_in[5];   // [H*HD, D]
    const float* ln2w = (const float*)d_in[6];
    const float* rw   = (const float*)d_in[7];   // [E, D]
    const float* wg   = (const float*)d_in[8];   // [E, D, F]
    const float* wu   = (const float*)d_in[9];
    const float* wd   = (const float*)d_in[10];  // [E, F, D]
    float* out = (float*)d_out;                  // [B,S,D] ++ [1] loss

    char* ws = (char*)d_ws;
    bf16*  h1    = (bf16*) (ws + OFF_H1);
    float* qb    = (float*)(ws + OFF_Q);
    float* kb    = (float*)(ws + OFF_K);
    float* vb    = (float*)(ws + OFF_V);
    bf16*  ctx   = (bf16*) (ws + OFF_CTX);
    bf16*  h2    = (bf16*) (ws + OFF_H2);
    float* gates = (float*)(ws + OFF_GATES);
    float* probs = (float*)(ws + OFF_PROBS);
    bf16*  act   = (bf16*) (ws + OFF_ACT);

    // ---- RMSNorm -> bf16 ----
    rmsnorm_kernel<<<Tc, 256, 0, stream>>>(x, ln1w, h1);

    // ---- Q/K/V projections (WMMA) ----
    gemm_bf16_kernel<<<dim3((Hc*HDc)/TILE_N, Tc/TILE_M), 256, 0, stream>>>(
        h1, wq, qb, nullptr, nullptr, 0, Tc, Hc*HDc, Dc, 0);
    gemm_bf16_kernel<<<dim3((KVHc*HDc)/TILE_N, Tc/TILE_M), 256, 0, stream>>>(
        h1, wk, kb, nullptr, nullptr, 0, Tc, KVHc*HDc, Dc, 0);
    gemm_bf16_kernel<<<dim3((KVHc*HDc)/TILE_N, Tc/TILE_M), 256, 0, stream>>>(
        h1, wv, vb, nullptr, nullptr, 0, Tc, KVHc*HDc, Dc, 0);

    // ---- RoPE ----
    rope_kernel<<<(Tc*Hc*32)/256,   256, 0, stream>>>(qb, Hc,   Tc*Hc*32);
    rope_kernel<<<(Tc*KVHc*32)/256, 256, 0, stream>>>(kb, KVHc, Tc*KVHc*32);

    // ---- causal attention (GQA), ctx in bf16 ----
    attention_kernel<<<(Bc*Hc*Sc)/8, 256, 0, stream>>>(qb, kb, vb, ctx);

    // ---- WO projection fused with residual, written straight into d_out ----
    gemm_bf16_kernel<<<dim3(Dc/TILE_N, Tc/TILE_M), 256, 0, stream>>>(
        ctx, wo, out, /*residual=*/x, nullptr, 0, Tc, Dc, Hc*HDc, 0);

    // ---- RMSNorm 2 (reads attn residual from d_out) ----
    rmsnorm_kernel<<<Tc, 256, 0, stream>>>(out, ln2w, h2);

    // ---- router ----
    router_kernel<<<Tc, 32, 0, stream>>>(h2, rw, gates, probs);

    // ---- experts: fused gate+up (WMMA) then scaled down-proj accumulate ----
    for (int e = 0; e < Ec; ++e) {
        moe_gateup_kernel<<<dim3(Fc/TILE_N, Tc/TILE_M), 256, 0, stream>>>(
            h2, wg + (size_t)e * Dc * Fc, wu + (size_t)e * Dc * Fc,
            act, Tc, Fc, Dc);
        gemm_bf16_kernel<<<dim3(Dc/TILE_N, Tc/TILE_M), 256, 0, stream>>>(
            act, wd + (size_t)e * Fc * Dc, out,
            nullptr, /*rowscale=*/gates + e, /*rs_stride=*/Ec,
            Tc, Dc, Fc, /*accumulate=*/1);
    }

    // ---- balance loss (deterministic reduction) ----
    balance_kernel<<<1, 256, 0, stream>>>(gates, probs, out + (size_t)Tc * Dc);
}
